// MultiheadSelfAttention_41369124995143
// MI455X (gfx1250) — matrix operand order repaired and checked by reading, not compiled, not run on hardware
//
#include <hip/hip_runtime.h>

typedef __attribute__((ext_vector_type(16))) __bf16 v16bf;
typedef __attribute__((ext_vector_type(8)))  __bf16 v8bf;
typedef __attribute__((ext_vector_type(4)))  __bf16 v4bf;
typedef __attribute__((ext_vector_type(8)))  float  v8f;
typedef __attribute__((ext_vector_type(4)))  unsigned int u32x4;
typedef __attribute__((ext_vector_type(8)))  int i32x8;
typedef __attribute__((ext_vector_type(4)))  int i32x4;

#define BB 4
#define TT 2048
#define MM 1024
#define DD 128
#define HH 8

#if __has_builtin(__builtin_amdgcn_tensor_load_to_lds) && \
    __has_builtin(__builtin_amdgcn_s_wait_tensorcnt)
#define HAVE_TDM 1
#else
#define HAVE_TDM 0
#endif

__device__ __forceinline__ v8f wmma_bf16(v16bf a, v16bf b, v8f c) {
    return __builtin_amdgcn_wmma_f32_16x16x32_bf16(
        /*neg_a=*/false, a, /*neg_b=*/false, b,
        /*c_mod=*/(short)0, c, /*reuse_a=*/false, /*reuse_b=*/false);
}

// 16-bit A-matrix 16x32: per lane the 16 values are one contiguous
// 16-element run [khalf*16 .. +15], matching the B fragments.
__device__ __forceinline__ v16bf afrag(const __bf16* rowk, int khalf) {
    return *(const v16bf*)(rowk + khalf * 16);
}

// ---------------------------------------------------------------------------
// Kernel 1: q/k/v = x @ w[h], RoPE in-register for q,k.
// Grid: (T/128, 3, B*H). 8 waves; each wave: 16 t-rows x full D=128.
// ---------------------------------------------------------------------------
__global__ __launch_bounds__(256)
void qkv_rope_kernel(const float* __restrict__ x,
                     const float* __restrict__ wq,
                     const float* __restrict__ wk,
                     const float* __restrict__ wv,
                     __bf16* __restrict__ qb,
                     __bf16* __restrict__ kb,
                     __bf16* __restrict__ vb) {
    __shared__ __attribute__((aligned(32))) __bf16 As[128 * 32];  // [t][k]
    __shared__ __attribute__((aligned(32))) __bf16 Bs[128 * 32];  // [d][k]

    const int tid = threadIdx.x;
    const int wave = tid >> 5, lane = tid & 31;
    const int lrow = lane & 15, khalf = lane >> 4;

    const int t0    = blockIdx.x * 128;
    const int which = blockIdx.y;
    const int bh    = blockIdx.z;
    const int b = bh / HH, h = bh % HH;

    const float* w = (which == 0) ? wq : (which == 1) ? wk : wv;
    __bf16* outp   = (which == 0) ? qb : (which == 1) ? kb : vb;

    v8f acc[8] = {};

    for (int k0 = 0; k0 < MM; k0 += 32) {
        // Stage x tile [128][32]: vectorized float4 loads -> bf16 quads
        #pragma unroll
        for (int i = 0; i < 4; ++i) {
            const int e = tid + i * 256;          // 1024 float4 chunks
            const int r = e >> 3, c4 = (e & 7) * 4;
            const float4 xx = *(const float4*)&x[(size_t)(b * TT + t0 + r) * MM + k0 + c4];
            v4bf p = {(__bf16)xx.x, (__bf16)xx.y, (__bf16)xx.z, (__bf16)xx.w};
            *(v4bf*)&As[r * 32 + c4] = p;
        }
        // Stage w tile transposed to [d][k] (contiguous-d loads, b16 scatter)
        #pragma unroll
        for (int i = 0; i < 4; ++i) {
            const int e = tid + i * 256;
            const int kk = e >> 5, d4 = (e & 31) * 4;
            const float4 ww = *(const float4*)&w[(size_t)(h * MM + k0 + kk) * DD + d4];
            Bs[(d4 + 0) * 32 + kk] = (__bf16)ww.x;
            Bs[(d4 + 1) * 32 + kk] = (__bf16)ww.y;
            Bs[(d4 + 2) * 32 + kk] = (__bf16)ww.z;
            Bs[(d4 + 3) * 32 + kk] = (__bf16)ww.w;
        }
        if (k0 + 32 < MM) {  // prefetch next K-step tiles into cache
            __builtin_prefetch(&x[(size_t)(b * TT + t0 + (tid >> 1)) * MM + k0 + 32 + (tid & 1) * 16], 0, 0);
            __builtin_prefetch(&w[(size_t)(h * MM + k0 + 32 + (tid >> 3)) * DD + (tid & 7) * 16], 0, 0);
        }
        __syncthreads();

        const v16bf a = afrag(&As[(wave * 16 + lrow) * 32], khalf);
        #pragma unroll
        for (int f = 0; f < 8; ++f) {
            const v16bf bf = *(const v16bf*)&Bs[(f * 16 + lrow) * 32 + khalf * 16];
            acc[f] = wmma_bf16(a, bf, acc[f]);
        }
        __syncthreads();
    }

    // RoPE: pairs (d, d+64) live in fragments f and f+4 at same element slot
    const int rbase = t0 + wave * 16;
    const size_t obase = (size_t)bh * TT * DD;
    if (which < 2) {
        #pragma unroll
        for (int f = 0; f < 4; ++f) {
            const int d = f * 16 + lrow;  // 0..63
            const float freq = __expf(-(float)d * (9.210340371976184f / 64.0f));
            #pragma unroll
            for (int v = 0; v < 8; ++v) {
                const int r = v + khalf * 8;
                const float th = (float)(rbase + r) * freq;
                const float c = __cosf(th), s = __sinf(th);
                const float ev = acc[f][v], od = acc[f + 4][v];
                acc[f][v]     = ev * c - od * s;
                acc[f + 4][v] = ev * s + od * c;
            }
        }
    }
    #pragma unroll
    for (int f = 0; f < 8; ++f) {
        const int d = f * 16 + lrow;
        #pragma unroll
        for (int v = 0; v < 8; ++v) {
            const int r = v + khalf * 8;
            outp[obase + (size_t)(rbase + r) * DD + d] = (__bf16)acc[f][v];
        }
    }
}

// ---------------------------------------------------------------------------
// Kernel 2: causal flash attention, online softmax. Grid: (T/128, 1, B*H).
// Workgroup-uniform j loop: K tile staged by the Tensor Data Mover (TDM),
// V tile staged cooperatively with an LDS transpose. Waves past their causal
// limit skip compute via a wave-uniform branch (EXEC all-1s at every WMMA).
// ---------------------------------------------------------------------------
__global__ __launch_bounds__(256)
void flash_attn_kernel(const __bf16* __restrict__ qb,
                       const __bf16* __restrict__ kb,
                       const __bf16* __restrict__ vb,
                       __bf16* __restrict__ ob) {
    __shared__ __attribute__((aligned(32))) __bf16 Ks[32 * 128];   // [j][d]
    __shared__ __attribute__((aligned(32))) __bf16 Vs[128 * 32];   // [d][j]
    __shared__ __attribute__((aligned(32))) __bf16 Ps[8][16 * 32]; // per-wave

    const int tid = threadIdx.x;
    const int wave = tid >> 5, lane = tid & 31;
    const int lrow = lane & 15, khalf = lane >> 4;

    const int qblk = blockIdx.x;
    const int bh   = blockIdx.z;
    const size_t base = (size_t)bh * TT * DD;
    const int qrow = qblk * 128 + wave * 16;

    // Resident Q fragments (4 K-chunks of 32 over D=128), vector loads
    const __bf16* qrp = qb + base + (size_t)(qrow + lrow) * DD;
    v16bf qf[4];
    #pragma unroll
    for (int c = 0; c < 4; ++c) qf[c] = afrag(qrp + c * 32, khalf);

    v8f O[8] = {};
    float mrow[8], lsum[8];
    #pragma unroll
    for (int v = 0; v < 8; ++v) { mrow[v] = -3.0e38f; lsum[v] = 0.0f; }

    const float scale = 1.0f / 128.0f;
    const int njb_wave = qrow / 32 + 1;     // this wave's causal j-block limit
    const int njb_max  = qblk * 4 + 4;      // workgroup-wide limit (wave 7)

    for (int jb = 0; jb < njb_max; ++jb) {
        const int j0 = jb * 32;

        // --- K tile (32x128 bf16, fully contiguous 8KB) -> LDS ---
#if HAVE_TDM
        if (wave == 0) {
            const __bf16* kp = kb + base + (size_t)j0 * DD;
            const unsigned long long ga = (unsigned long long)(uintptr_t)kp;
            const unsigned ldsa = (unsigned)(uintptr_t)&Ks[0];
            // D# group0: count=1 | lds_addr | global_addr | type=2
            u32x4 g0 = {1u, ldsa, (unsigned)ga,
                        (unsigned)((ga >> 32) & 0x01ffffffu) | (2u << 30)};
            // D# group1: data_size=2B; 1D tile of 4096 elems; stride0=4096
            i32x8 g1 = {(int)(1u << 16),
                        (int)((4096u & 0xffffu) << 16),   // tensor_dim0 lo16
                        (int)(1u << 16),                  // tensor_dim1=1
                        (int)(4096u << 16),               // tile_dim0=4096
                        0,                                // tile_dim1/2 unused
                        (int)4096u,                       // tensor_dim0_stride
                        0, 0};
            i32x4 z4 = {0, 0, 0, 0};
#if __clang_major__ >= 23
            i32x8 z8 = {0, 0, 0, 0, 0, 0, 0, 0};
            __builtin_amdgcn_tensor_load_to_lds(g0, g1, z4, z4, z8, 0);
#else
            __builtin_amdgcn_tensor_load_to_lds(g0, g1, z4, z4, 0);
#endif
        }
#else
        #pragma unroll
        for (int i = 0; i < 2; ++i) {
            const int e = tid + i * 256;  // 512 chunks of 8 bf16
            *(v8bf*)&Ks[e * 8] = *(const v8bf*)&kb[base + (size_t)j0 * DD + e * 8];
        }
#endif
        // --- V tile -> LDS transposed [d][j] (b128 loads, b16 scatter) ---
        #pragma unroll
        for (int i = 0; i < 2; ++i) {
            const int e = tid + i * 256;  // 512 chunks of 8
            const int j = e >> 4, c8 = (e & 15) * 8;
            const v8bf vv = *(const v8bf*)&vb[base + (size_t)(j0 + j) * DD + c8];
            #pragma unroll
            for (int u = 0; u < 8; ++u) Vs[(c8 + u) * 32 + j] = vv[u];
        }
#if HAVE_TDM
        if (wave == 0) __builtin_amdgcn_s_wait_tensorcnt(0);
#endif
        __syncthreads();

        if (jb < njb_wave) {   // wave-uniform causal skip
            // S = Q K^T over 32 j-cols (two 16-col C tiles)
            v8f S0 = {}, S1 = {};
            #pragma unroll
            for (int c = 0; c < 4; ++c) {
                const v16bf b0 = *(const v16bf*)&Ks[lrow * 128 + c * 32 + khalf * 16];
                const v16bf b1 = *(const v16bf*)&Ks[(16 + lrow) * 128 + c * 32 + khalf * 16];
                S0 = wmma_bf16(qf[c], b0, S0);
                S1 = wmma_bf16(qf[c], b1, S1);
            }

            // Mask + scale + online softmax (row = v + 8*khalf)
            float alpha[8], p0v[8], p1v[8];
            #pragma unroll
            for (int v = 0; v < 8; ++v) {
                const int i  = qrow + v + khalf * 8;
                const int jA = j0 + lrow;
                const int jB = j0 + 16 + lrow;
                float s0 = (jA <= i) ? S0[v] * scale : -1.0e30f;
                float s1 = (jB <= i) ? S1[v] * scale : -1.0e30f;
                float mx = fmaxf(s0, s1);
                #pragma unroll
                for (int m = 1; m < 16; m <<= 1) mx = fmaxf(mx, __shfl_xor(mx, m, 32));
                const float mnew = fmaxf(mrow[v], mx);
                const float a  = __expf(mrow[v] - mnew);
                const float p0 = __expf(s0 - mnew);
                const float p1 = __expf(s1 - mnew);
                float rs = p0 + p1;
                #pragma unroll
                for (int m = 1; m < 16; m <<= 1) rs += __shfl_xor(rs, m, 32);
                lsum[v] = lsum[v] * a + rs;
                mrow[v] = mnew;
                alpha[v] = a; p0v[v] = p0; p1v[v] = p1;
            }
            #pragma unroll
            for (int f = 0; f < 8; ++f)
                #pragma unroll
                for (int v = 0; v < 8; ++v) O[f][v] *= alpha[v];

            // P: C-layout regs -> LDS (16x32) -> A fragment
            #pragma unroll
            for (int v = 0; v < 8; ++v) {
                const int r = v + khalf * 8;
                Ps[wave][r * 32 + lrow]      = (__bf16)p0v[v];
                Ps[wave][r * 32 + 16 + lrow] = (__bf16)p1v[v];
            }
            const v16bf pf = afrag(&Ps[wave][lrow * 32], khalf);

            // O += P V : V B-fragments are contiguous in transposed Vs
            #pragma unroll
            for (int f = 0; f < 8; ++f) {
                const int dcol = f * 16 + lrow;
                const v16bf vf = *(const v16bf*)&Vs[dcol * 32 + khalf * 16];
                O[f] = wmma_bf16(pf, vf, O[f]);
            }
        }
        __syncthreads();
    }

    #pragma unroll
    for (int v = 0; v < 8; ++v) lsum[v] = 1.0f / lsum[v];
    #pragma unroll
    for (int f = 0; f < 8; ++f) {
        const int d = f * 16 + lrow;
        #pragma unroll
        for (int v = 0; v < 8; ++v) {
            const int r = qrow + v + khalf * 8;
            ob[base + (size_t)r * DD + d] = (__bf16)(O[f][v] * lsum[v]);
        }
    }
}

// ---------------------------------------------------------------------------
// Kernel 3: r[b,t,m] = sum_hd o[b,t,hd] * wo_flat[hd,m]. Grid (T/128,M/128,B).
// ---------------------------------------------------------------------------
__global__ __launch_bounds__(256)
void out_proj_kernel(const __bf16* __restrict__ ob,
                     const float* __restrict__ wo,
                     float* __restrict__ out) {
    __shared__ __attribute__((aligned(32))) __bf16 As[128 * 32];  // [t][kk]
    __shared__ __attribute__((aligned(32))) __bf16 Bs[128 * 32];  // [m][kk]

    const int tid = threadIdx.x;
    const int wave = tid >> 5, lane = tid & 31;
    const int lrow = lane & 15, khalf = lane >> 4;

    const int t0 = blockIdx.x * 128;
    const int m0 = blockIdx.y * 128;
    const int b  = blockIdx.z;

    v8f acc[8] = {};

    for (int k0 = 0; k0 < MM; k0 += 32) {
        // A tile: o is already bf16; 16-byte vector copies (head-aware)
        #pragma unroll
        for (int i = 0; i < 2; ++i) {
            const int e = tid + i * 256;  // 512 chunks of 8
            const int r = e >> 2, c8 = (e & 3) * 8;
            const int hd = k0 + c8;
            const int h = hd >> 7, d = hd & 127;
            *(v8bf*)&As[r * 32 + c8] =
                *(const v8bf*)&ob[((size_t)(b * HH + h) * TT + t0 + r) * DD + d];
        }
        // B tile: wo fp32 -> bf16, transposed to [m][kk]
        #pragma unroll
        for (int i = 0; i < 4; ++i) {
            const int e = tid + i * 256;
            const int kk = e >> 5, m4 = (e & 31) * 4;
            const float4 ww = *(const float4*)&wo[(size_t)(k0 + kk) * MM + m0 + m4];
            Bs[(m4 + 0) * 32 + kk] = (__bf16)ww.x;
            Bs[(m4 + 1) * 32 + kk] = (__bf16)ww.y;
            Bs[(m4 + 2) * 32 + kk] = (__bf16)ww.z;
            Bs[(m4 + 3) * 32 + kk] = (__bf16)ww.w;
        }
        if (k0 + 32 < MM)
            __builtin_prefetch(&wo[(size_t)(k0 + 32 + (tid >> 3)) * MM + m0 + (tid & 7) * 16], 0, 0);
        __syncthreads();

        const v16bf a = afrag(&As[(wave * 16 + lrow) * 32], khalf);
        #pragma unroll
        for (int f = 0; f < 8; ++f) {
            const v16bf bf = *(const v16bf*)&Bs[(f * 16 + lrow) * 32 + khalf * 16];
            acc[f] = wmma_bf16(a, bf, acc[f]);
        }
        __syncthreads();
    }

    #pragma unroll
    for (int f = 0; f < 8; ++f) {
        const int m = m0 + f * 16 + lrow;
        #pragma unroll
        for (int v = 0; v < 8; ++v) {
            const int r = t0 + wave * 16 + v + khalf * 8;
            out[((size_t)b * TT + r) * MM + m] = acc[f][v];
        }
    }
}

// ---------------------------------------------------------------------------
extern "C" void kernel_launch(void* const* d_in, const int* in_sizes, int n_in,
                              void* d_out, int out_size, void* d_ws, size_t ws_size,
                              hipStream_t stream) {
    const float* x  = (const float*)d_in[0];
    const float* wq = (const float*)d_in[1];
    const float* wk = (const float*)d_in[2];
    const float* wv = (const float*)d_in[3];
    const float* wo = (const float*)d_in[4];
    float* out = (float*)d_out;

    const size_t nqkv = (size_t)BB * HH * TT * DD;   // 8.39M bf16 elems each
    __bf16* qb   = (__bf16*)d_ws;
    __bf16* kb   = qb + nqkv;
    __bf16* vb   = kb + nqkv;
    __bf16* obuf = vb + nqkv;                        // ~67 MB total

    dim3 blk(256, 1, 1);
    qkv_rope_kernel<<<dim3(TT / 128, 3, BB * HH), blk, 0, stream>>>(
        x, wq, wk, wv, qb, kb, vb);
    flash_attn_kernel<<<dim3(TT / 128, 1, BB * HH), blk, 0, stream>>>(
        qb, kb, vb, obuf);
    out_proj_kernel<<<dim3(TT / 128, MM / 128, BB), blk, 0, stream>>>(
        obuf, wo, out);
}
